// ANNMCModel_17721035063863
// MI455X (gfx1250) — matrix operand order repaired and checked by reading, not compiled, not run on hardware
//
#include <hip/hip_runtime.h>
#include <hip/hip_bf16.h>

typedef __attribute__((ext_vector_type(16))) __bf16 v16bf;
typedef __attribute__((ext_vector_type(8)))  __bf16 v8bf;
typedef __attribute__((ext_vector_type(8)))  float  v8f;

// ---------------------------------------------------------------------------
// WMMA helper: D = A(16x32 bf16) * B(32x16 bf16) + C(16x16 f32)
// ---------------------------------------------------------------------------
__device__ __forceinline__ v8f wmma_bf16(v16bf a, v16bf b, v8f c) {
    return __builtin_amdgcn_wmma_f32_16x16x32_bf16(
        /*neg_a=*/false, a, /*neg_b=*/false, b,
        /*c_mod=*/(short)0, c, /*reuse_a=*/false, /*reuse_b=*/false);
}

// A-fragment lane mapping (ISA 7.12.2, 16-bit A 16x32):
//   lane l: m = l&15, kh = l>>4
//   element e<8  -> K = k0 + 8*kh + e          (8 consecutive)
//   element e>=8 -> K = k0 + 16 + 8*kh + (e-8) (8 consecutive)
__device__ __forceinline__ v16bf load_a_f32(const float* __restrict__ row,
                                            int k0, int kh) {
    const float* p = row + k0 + 8 * kh;
    float4 f0 = *(const float4*)(p);
    float4 f1 = *(const float4*)(p + 4);
    float4 f2 = *(const float4*)(p + 16);
    float4 f3 = *(const float4*)(p + 20);
    v16bf a;
    a[0]=(__bf16)f0.x; a[1]=(__bf16)f0.y; a[2]=(__bf16)f0.z; a[3]=(__bf16)f0.w;
    a[4]=(__bf16)f1.x; a[5]=(__bf16)f1.y; a[6]=(__bf16)f1.z; a[7]=(__bf16)f1.w;
    a[8]=(__bf16)f2.x; a[9]=(__bf16)f2.y; a[10]=(__bf16)f2.z; a[11]=(__bf16)f2.w;
    a[12]=(__bf16)f3.x; a[13]=(__bf16)f3.y; a[14]=(__bf16)f3.z; a[15]=(__bf16)f3.w;
    return a;
}

__device__ __forceinline__ v16bf load_a_lds_f32(const float* S, int ldk,
                                                int m, int k0, int kh) {
    return load_a_f32(S + m * ldk, k0, kh);
}

__device__ __forceinline__ v16bf load_a_lds_bf16(const __bf16* S, int ldk,
                                                 int m, int k0, int kh) {
    const __bf16* p = S + m * ldk + k0 + 8 * kh;
    v8bf lo = *(const v8bf*)(p);        // 16B-aligned (k0 mult of 32, 8*kh)
    v8bf hi = *(const v8bf*)(p + 16);
    v16bf a;
#pragma unroll
    for (int e = 0; e < 8; ++e) { a[e] = lo[e]; a[8 + e] = hi[e]; }
    return a;
}

// ---------------------------------------------------------------------------
// Pre-swizzle a KxN fp32 weight into padded bf16 B-fragment layout:
//   frag[(((kt*NT)+nt)*32 + lane)*16 + e] = W[kt*32 + (lane>>4)*8 + (e&7) + (e>>3)*16][nt*16 + (lane&15)]
// (ISA 7.12.2: B 32x16, lanes 0-15 hold K=0..7,16..23, lanes 16-31 hold K=8..15,24..31)
// ---------------------------------------------------------------------------
__global__ void fill_frag_kernel(__bf16* __restrict__ dst,
                                 const float* __restrict__ src,
                                 int Kreal, int Nreal, int KT, int NT) {
    int t = blockIdx.x * blockDim.x + threadIdx.x;
    int total = KT * NT * 512;
    if (t >= total) return;
    int e    = t & 15;
    int lane = (t >> 4) & 31;
    int nt   = (t >> 9) % NT;
    int kt   = t / (512 * NT);
    int col  = nt * 16 + (lane & 15);
    int k    = kt * 32 + (lane >> 4) * 8 + (e & 7) + ((e >> 3) << 4);
    float v = (k < Kreal && col < Nreal) ? src[(size_t)k * Nreal + col] : 0.0f;
    dst[t] = (__bf16)v;
}

// ---------------------------------------------------------------------------
// Kernel 1: fused ann1.  One BLOCK (4 waves) computes 16 rows of:
//   h1 = relu(x(16x3072) @ w1a(3072x100)+b1a)      [N pad 112; K split 4 ways]
//   h2 = relu(h1 @ w1b(100x25)+b1b)                [K pad 128, N pad 32; wave 0]
//   x2 = h2 @ w1c(25x1536)+b1c                     [K pad 32; 96 N-tiles / 4 waves]
// K-split over waves -> 2048 independent streaming waves for the 100 MB x read.
// ---------------------------------------------------------------------------
__global__ __launch_bounds__(128) void ann1_kernel(
    const float* __restrict__ x,
    const __bf16* __restrict__ w1a_f, const float* __restrict__ b1a,
    const __bf16* __restrict__ w1b_f, const float* __restrict__ b1b,
    const __bf16* __restrict__ w1c_f, const float* __restrict__ b1c,
    float* __restrict__ x2) {
    __shared__ float part[4][16][112];   // 28 KB: per-wave partial sums
    __shared__ float Smem[16][128];      //  8 KB: staged activations
    const int wave = threadIdx.x >> 5;
    const int lane = threadIdx.x & 31;
    const int m    = lane & 15;
    const int kh   = lane >> 4;
    const int m0   = blockIdx.x * 16;
    float* S = &Smem[0][0];

    // ---- stage A: x @ w1a, K=3072; wave w takes kt = w, w+4, ...
    v8f acc[7];
#pragma unroll
    for (int nt = 0; nt < 7; ++nt)
#pragma unroll
        for (int i = 0; i < 8; ++i) acc[nt][i] = 0.0f;
    const float* xrow = x + (size_t)(m0 + m) * 3072;
    for (int kt = wave; kt < 96; kt += 4) {
        v16bf a = load_a_f32(xrow, kt * 32, kh);
#pragma unroll
        for (int nt = 0; nt < 7; ++nt) {
            v16bf b = *(const v16bf*)(w1a_f + ((size_t)(kt * 7 + nt) * 32 + lane) * 16);
            acc[nt] = wmma_bf16(a, b, acc[nt]);
        }
    }
    // C/D layout: element i, lane l -> M = i + 8*(l>>4), N = l&15
#pragma unroll
    for (int nt = 0; nt < 7; ++nt)
#pragma unroll
        for (int i = 0; i < 8; ++i)
            part[wave][i + 8 * kh][nt * 16 + m] = acc[nt][i];
    __syncthreads();

    // reduce 4 partials + bias + relu -> S[16][128] (cols 112..127 zeroed)
    for (int idx = threadIdx.x; idx < 16 * 128; idx += 128) {
        int row = idx >> 7, col = idx & 127;
        float v = 0.0f;
        if (col < 112) {
            v = part[0][row][col] + part[1][row][col] +
                part[2][row][col] + part[3][row][col];
            if (col < 100) v += b1a[col];
            v = fmaxf(v, 0.0f);
        }
        S[idx] = v;
    }
    __syncthreads();

    // ---- stage B (wave 0 only): h1 @ w1b, K=128 (4 steps), N=32 (real 25)
    if (wave == 0) {
        v8f acc2[2];
#pragma unroll
        for (int nt = 0; nt < 2; ++nt) {
            int n = nt * 16 + m;
            float bv = (n < 25) ? b1b[n] : 0.0f;
#pragma unroll
            for (int i = 0; i < 8; ++i) acc2[nt][i] = bv;
        }
#pragma unroll
        for (int kt = 0; kt < 4; ++kt) {
            v16bf a = load_a_lds_f32(S, 128, m, kt * 32, kh);
#pragma unroll
            for (int nt = 0; nt < 2; ++nt) {
                v16bf b = *(const v16bf*)(w1b_f + ((size_t)(kt * 2 + nt) * 32 + lane) * 16);
                acc2[nt] = wmma_bf16(a, b, acc2[nt]);
            }
        }
#pragma unroll
        for (int nt = 0; nt < 2; ++nt)
#pragma unroll
            for (int i = 0; i < 8; ++i)
                S[(i + 8 * kh) * 128 + nt * 16 + m] = fmaxf(acc2[nt][i], 0.0f);
    }
    __syncthreads();

    // ---- stage C: h2 @ w1c, K=32 (real 25), 96 N-tiles split over 4 waves
    {
        v16bf a = load_a_lds_f32(S, 128, m, 0, kh);
        for (int nt = wave; nt < 96; nt += 4) {
            v16bf b = *(const v16bf*)(w1c_f + ((size_t)nt * 32 + lane) * 16);
            v8f c;
            float bv = b1c[nt * 16 + m];
#pragma unroll
            for (int i = 0; i < 8; ++i) c[i] = bv;
            c = wmma_bf16(a, b, c);
#pragma unroll
            for (int i = 0; i < 8; ++i)
                x2[(size_t)(m0 + i + 8 * kh) * 1536 + nt * 16 + m] = c[i];
        }
    }
}

// ---------------------------------------------------------------------------
// Kernel 2: state quantization + TPM gather + fused ann2
// One wave handles 16 flattened rows r = b*24 + j.
// ---------------------------------------------------------------------------
__device__ __forceinline__ int state_of(float v) {
    int s = (int)rintf((v + 5.0f) / 0.1f);
    return s < 0 ? 0 : (s > 98 ? 98 : s);
}
__device__ __forceinline__ float sanit(float v) { return (v == v) ? v : 0.0f; }

__global__ __launch_bounds__(128) void gather_ann2_kernel(
    const float* __restrict__ x, const float* __restrict__ x2,
    const float* __restrict__ TPM,
    const __bf16* __restrict__ w2a_f, const float* __restrict__ b2a,
    const __bf16* __restrict__ w2b_f, const float* __restrict__ b2b,
    const __bf16* __restrict__ w2c_f, const float* __restrict__ b2c,
    float* __restrict__ out) {
    __shared__ __bf16 tenS[4][16][384];   // 48 KB
    __shared__ float  hS[4][16][32];      //  8 KB
    const int wave = threadIdx.x >> 5;
    const int lane = threadIdx.x & 31;
    const int m    = lane & 15;
    const int kh   = lane >> 4;
    const int r0   = (blockIdx.x * 4 + wave) * 16;
    __bf16* T = &tenS[wave][0][0];
    float*  H = &hS[wave][0][0];

    // ---- build ten(16 x 384) in LDS as bf16
    for (int idx = lane; idx < 1024; idx += 32) {
        int rr = idx >> 6, k = idx & 63;
        int r  = r0 + rr;
        int bb = r / 24, j = r - bb * 24;
        float x2v = sanit(x2[(size_t)r * 64 + k]);
        float pv  = sanit(j ? x2[(size_t)(r - 1) * 64 + k]
                            : x[(size_t)bb * 3072 + 47 * 64 + k]);
        int s1 = state_of(pv), s2 = state_of(x2v);
        const float* tb = TPM + ((size_t)k * 99 + s1) * 99;
        float g2 = sanit(tb[s2]);
        float g3 = sanit(tb[s1 + 1 > 98 ? 98 : s1 + 1]);
        float g4 = sanit(tb[s1 - 1 < 0 ? 0 : s1 - 1]);
        float g5 = sanit(tb[s1 + 2 > 98 ? 98 : s1 + 2]);
        float g6 = sanit(tb[s1 - 2 < 0 ? 0 : s1 - 2]);
        __bf16* t = T + rr * 384 + 6 * k;
        t[0] = (__bf16)x2v; t[1] = (__bf16)g2; t[2] = (__bf16)g3;
        t[3] = (__bf16)g4;  t[4] = (__bf16)g5; t[5] = (__bf16)g6;
    }

    // ---- stage 2a: ten @ w2a, K=384 (12 steps), N=32 (2 tiles, real 20)
    v8f acc[2];
#pragma unroll
    for (int nt = 0; nt < 2; ++nt) {
        int n = nt * 16 + m;
        float bv = (n < 20) ? b2a[n] : 0.0f;
#pragma unroll
        for (int i = 0; i < 8; ++i) acc[nt][i] = bv;
    }
    for (int kt = 0; kt < 12; ++kt) {
        v16bf a = load_a_lds_bf16(T, 384, m, kt * 32, kh);
#pragma unroll
        for (int nt = 0; nt < 2; ++nt) {
            v16bf b = *(const v16bf*)(w2a_f + ((size_t)(kt * 2 + nt) * 32 + lane) * 16);
            acc[nt] = wmma_bf16(a, b, acc[nt]);
        }
    }
#pragma unroll
    for (int nt = 0; nt < 2; ++nt)
#pragma unroll
        for (int i = 0; i < 8; ++i)
            H[(i + 8 * kh) * 32 + nt * 16 + m] = fmaxf(acc[nt][i], 0.0f);

    // ---- stage 2b: K=32 (real 20), N=16 (real 10)
    {
        v16bf a = load_a_lds_f32(H, 32, m, 0, kh);
        v8f c;
        float bv = (m < 10) ? b2b[m] : 0.0f;
#pragma unroll
        for (int i = 0; i < 8; ++i) c[i] = bv;
        v16bf b = *(const v16bf*)(w2b_f + (size_t)lane * 16);
        c = wmma_bf16(a, b, c);
#pragma unroll
        for (int i = 0; i < 8; ++i) {
            H[(i + 8 * kh) * 32 + m]      = fmaxf(c[i], 0.0f);
            H[(i + 8 * kh) * 32 + 16 + m] = 0.0f;   // K-pad for next stage
        }
    }

    // ---- stage 2c: K=32 (real 10), N=64 (4 tiles) -> out
    {
        v16bf a = load_a_lds_f32(H, 32, m, 0, kh);
#pragma unroll
        for (int nt = 0; nt < 4; ++nt) {
            v16bf b = *(const v16bf*)(w2c_f + ((size_t)nt * 32 + lane) * 16);
            v8f c;
            float bv = b2c[nt * 16 + m];
#pragma unroll
            for (int i = 0; i < 8; ++i) c[i] = bv;
            c = wmma_bf16(a, b, c);
#pragma unroll
            for (int i = 0; i < 8; ++i)
                out[(size_t)(r0 + i + 8 * kh) * 64 + nt * 16 + m] = c[i];
        }
    }
}

// ---------------------------------------------------------------------------
extern "C" void kernel_launch(void* const* d_in, const int* in_sizes, int n_in,
                              void* d_out, int out_size, void* d_ws, size_t ws_size,
                              hipStream_t stream) {
    const float* x   = (const float*)d_in[0];
    const float* TPM = (const float*)d_in[1];
    const float* w1a = (const float*)d_in[2];
    const float* b1a = (const float*)d_in[3];
    const float* w1b = (const float*)d_in[4];
    const float* b1b = (const float*)d_in[5];
    const float* w1c = (const float*)d_in[6];
    const float* b1c = (const float*)d_in[7];
    const float* w2a = (const float*)d_in[8];
    const float* b2a = (const float*)d_in[9];
    const float* w2b = (const float*)d_in[10];
    const float* b2b = (const float*)d_in[11];
    const float* w2c = (const float*)d_in[12];
    const float* b2c = (const float*)d_in[13];

    char* ws = (char*)d_ws;
    size_t off = 0;
    float* x2 = (float*)(ws + off);      off += (size_t)8192 * 1536 * 4;   // 50.3 MB
    __bf16* w1a_f = (__bf16*)(ws + off); off += (size_t)96 * 7 * 512 * 2;  // 672 KB
    __bf16* w1b_f = (__bf16*)(ws + off); off += (size_t)4 * 2 * 512 * 2;
    __bf16* w1c_f = (__bf16*)(ws + off); off += (size_t)1 * 96 * 512 * 2;
    __bf16* w2a_f = (__bf16*)(ws + off); off += (size_t)12 * 2 * 512 * 2;
    __bf16* w2b_f = (__bf16*)(ws + off); off += (size_t)1 * 1 * 512 * 2;
    __bf16* w2c_f = (__bf16*)(ws + off); off += (size_t)1 * 4 * 512 * 2;

    auto fill = [&](__bf16* dst, const float* src, int K, int N, int KT, int NT) {
        int total = KT * NT * 512;
        fill_frag_kernel<<<(total + 255) / 256, 256, 0, stream>>>(dst, src, K, N, KT, NT);
    };
    fill(w1a_f, w1a, 3072, 100, 96, 7);
    fill(w1b_f, w1b,  100,  25,  4, 2);
    fill(w1c_f, w1c,   25, 1536, 1, 96);
    fill(w2a_f, w2a,  384,  20, 12, 2);
    fill(w2b_f, w2b,   20,  10,  1, 1);
    fill(w2c_f, w2c,   10,  64,  1, 4);

    // one 16-row tile per block, K-split over its 4 waves -> 2048 waves
    ann1_kernel<<<512, 128, 0, stream>>>(x, w1a_f, b1a, w1b_f, b1b, w1c_f, b1c, x2);
    // 196608/16 = 12288 row-tiles, 4 waves/block
    gather_ann2_kernel<<<3072, 128, 0, stream>>>(x, x2, TPM, w2a_f, b2a,
                                                 w2b_f, b2b, w2c_f, b2c,
                                                 (float*)d_out);
}
